// Attention_40767829573686
// MI455X (gfx1250) — compile-verified, matrix operand order checked by reading, not run on hardware
//
#include <hip/hip_runtime.h>

// ---------------------------------------------------------------------------
// MI455X (gfx1250) fused QKV-projection + flash-attention, bf16 WMMA path.
//   Kernel 0: one-shot convert Wq/Wk/Wv (f32, [k][j]) -> bf16 W^T [which][j][k]
//   Kernel 1: Q/K/V = x@W + b   (bf16; V stored transposed [BH,D,T]).
//   Kernel 2: flash attention, S^T = K*Q^T trick (exp(S^T) C-layout fragments
//             feed the PV matmul as B operands with no cross-lane transpose).
//             K/V tiles staged in LDS via the Tensor Data Mover (TDM),
//             double-buffered, shared by all 8 waves; 2 q-tiles per wave.
// ---------------------------------------------------------------------------

typedef __attribute__((ext_vector_type(16))) __bf16 v16bf;
typedef __attribute__((ext_vector_type(8)))  __bf16 bf16x8;
typedef __attribute__((ext_vector_type(8)))  float  v8f;
typedef __attribute__((ext_vector_type(4)))  unsigned int u32x4;
typedef __attribute__((ext_vector_type(8)))  int i32x8;
typedef __attribute__((ext_vector_type(4)))  int i32x4;

#define T_SEQ  4096
#define BATCH  2
#define NHEADS 8
#define HDIM   64
#define MDIM   512
#define W_ELEMS (MDIM * MDIM)
#define KV_TILE 128                       // kv rows staged per TDM transfer

// ---------------------------------------------------------------------------
// TDM: issue a 2-D tile load (global -> LDS). data_size = 2 bytes (bf16).
// Descriptor packing per CDNA5 ISA 8.3/8.4 (D# groups 0/1; groups 2/3 benign).
// This toolchain exposes the 6-arg builtin: (g0, g1, g2, g3, extra, cpol).
// ---------------------------------------------------------------------------
__device__ __forceinline__ void tdm_load_tile_2d(
    unsigned lds_off, const void* gaddr,
    unsigned tensor_d0, unsigned tensor_d1,
    unsigned tile_d0, unsigned tile_d1,
    unsigned long long stride0)
{
  unsigned long long ga = (unsigned long long)(size_t)gaddr;
  u32x4 g0;
  g0[0] = 1u;                                            // count=1, user mode
  g0[1] = lds_off;                                       // lds_addr (bytes)
  g0[2] = (unsigned)(ga & 0xFFFFFFFFu);                  // global_addr[31:0]
  g0[3] = (unsigned)((ga >> 32) & 0x01FFFFFFu)           // global_addr[56:32]
        | (2u << 30);                                    // type=2 ("image")
  i32x8 g1;
  g1[0] = (int)(1u << 16);                               // data_size=1 -> 2B
  g1[1] = (int)((tensor_d0 & 0xFFFFu) << 16);            // tensor_dim0 lo16
  g1[2] = (int)(((tensor_d0 >> 16) & 0xFFFFu) |
                ((tensor_d1 & 0xFFFFu) << 16));          // d0 hi16 | d1 lo16
  g1[3] = (int)(((tensor_d1 >> 16) & 0xFFFFu) |
                ((tile_d0 & 0xFFFFu) << 16));            // d1 hi16 | tile_dim0
  g1[4] = (int)(tile_d1 & 0xFFFFu);                      // tile_dim1, tile_dim2=0
  g1[5] = (int)(unsigned)(stride0 & 0xFFFFFFFFull);      // tensor_dim0_stride lo
  g1[6] = (int)(unsigned)((stride0 >> 32) & 0xFFFFull);  // stride0 hi, stride1=0
  g1[7] = 0;
  i32x4 g2; g2[0] = 1; g2[1] = 0; g2[2] = 0; g2[3] = 0;  // tensor_dim2=1
  i32x4 g3; g3[0] = 0; g3[1] = 0; g3[2] = 0; g3[3] = 0;
  i32x8 g4;
  g4[0] = 0; g4[1] = 0; g4[2] = 0; g4[3] = 0;
  g4[4] = 0; g4[5] = 0; g4[6] = 0; g4[7] = 0;
  __builtin_amdgcn_tensor_load_to_lds(g0, g1, g2, g3, g4, 0);
}

__device__ __forceinline__ unsigned lds_offset_of(const void* p) {
  return (unsigned)(size_t)p;        // LDS aperture: addr[31:0] == LDS offset
}

// ---------------------------------------------------------------------------
// Kernel 0: W (f32 [k][j]) -> Wt (bf16 [which][j][k]); one-shot, 3 MB.
// ---------------------------------------------------------------------------
__global__ __launch_bounds__(256) void prep_w_kernel(
    const float* __restrict__ Wq, const float* __restrict__ Wk,
    const float* __restrict__ Wv, __bf16* __restrict__ Wt)
{
  const int id    = blockIdx.x * 256 + threadIdx.x;
  const int which = id >> 16;
  const int r     = id & 0xFFFF;
  const int k     = r >> 7;
  const int j4    = (r & 127) * 4;
  const float* W  = which == 0 ? Wq : (which == 1 ? Wk : Wv);

  float4 w = *(const float4*)(W + (size_t)k * MDIM + j4);
  __bf16* dst = Wt + (size_t)which * W_ELEMS;
  dst[(size_t)(j4 + 0) * MDIM + k] = (__bf16)w.x;
  dst[(size_t)(j4 + 1) * MDIM + k] = (__bf16)w.y;
  dst[(size_t)(j4 + 2) * MDIM + k] = (__bf16)w.z;
  dst[(size_t)(j4 + 3) * MDIM + k] = (__bf16)w.w;
}

// ---------------------------------------------------------------------------
// Kernel 1: QKV projection. grid = (B*T/128, NHEADS, 3), block = 256 (8 waves).
// ---------------------------------------------------------------------------
__global__ __launch_bounds__(256) void qkv_proj_kernel(
    const float* __restrict__ x,
    const float* __restrict__ bq, const float* __restrict__ bk,
    const float* __restrict__ bv, const __bf16* __restrict__ WtAll,
    __bf16* __restrict__ Qb, __bf16* __restrict__ Kb, __bf16* __restrict__ Vt)
{
  const int which = blockIdx.z;                    // 0=Q 1=K 2=V
  const __bf16* Wt  = WtAll + (size_t)which * W_ELEMS;
  const float* bias = which == 0 ? bq : (which == 1 ? bk : bv);

  const int tid  = threadIdx.x;
  const int wave = tid >> 5;
  const int lane = tid & 31;
  const int l16  = lane & 15;
  const int hl   = lane >> 4;

  const int h     = blockIdx.y;
  const int jbase = h * HDIM;
  const int nrow0 = blockIdx.x * 128 + wave * 16;
  const int b     = nrow0 / T_SEQ;
  const int t0    = nrow0 - b * T_SEQ;
  const int bh    = b * NHEADS + h;

  const v8f zero8 = {0.f,0.f,0.f,0.f,0.f,0.f,0.f,0.f};
  v8f acc[4] = {zero8, zero8, zero8, zero8};

  const float*  xrow = x  + (size_t)(nrow0 + l16) * MDIM;
  const __bf16* wrow = Wt + (size_t)(jbase + l16) * MDIM;

  for (int k0 = 0; k0 < MDIM; k0 += 32) {
    __builtin_prefetch(xrow + k0 + 32, 0, 1);
    __builtin_prefetch(wrow + k0 + 32, 0, 1);

    const float* xr = xrow + k0 + 8 * hl;
    float4 a0 = *(const float4*)(xr);
    float4 a1 = *(const float4*)(xr + 4);
    float4 a2 = *(const float4*)(xr + 16);
    float4 a3 = *(const float4*)(xr + 20);
    v16bf afrag;
    afrag[0]=(__bf16)a0.x;  afrag[1]=(__bf16)a0.y;
    afrag[2]=(__bf16)a0.z;  afrag[3]=(__bf16)a0.w;
    afrag[4]=(__bf16)a1.x;  afrag[5]=(__bf16)a1.y;
    afrag[6]=(__bf16)a1.z;  afrag[7]=(__bf16)a1.w;
    afrag[8]=(__bf16)a2.x;  afrag[9]=(__bf16)a2.y;
    afrag[10]=(__bf16)a2.z; afrag[11]=(__bf16)a2.w;
    afrag[12]=(__bf16)a3.x; afrag[13]=(__bf16)a3.y;
    afrag[14]=(__bf16)a3.z; afrag[15]=(__bf16)a3.w;

    #pragma unroll
    for (int c = 0; c < 4; ++c) {
      const __bf16* wr = wrow + (size_t)(16 * c) * MDIM + k0 + 8 * hl;
      bf16x8 blo = *(const bf16x8*)(wr);
      bf16x8 bhi = *(const bf16x8*)(wr + 16);
      v16bf bfrag;
      #pragma unroll
      for (int e = 0; e < 8; ++e) { bfrag[e] = blo[e]; bfrag[e + 8] = bhi[e]; }
      acc[c] = __builtin_amdgcn_wmma_f32_16x16x32_bf16(
          false, afrag, false, bfrag, (short)0, acc[c], false, false);
    }
  }

  #pragma unroll
  for (int c = 0; c < 4; ++c) {
    const int   dcol = 16 * c + l16;
    const float bv_  = bias[jbase + dcol];
    const int   trow = t0 + 8 * hl;
    if (which == 2) {                               // V transposed: [BH][D][T]
      bf16x8 pk;
      #pragma unroll
      for (int j = 0; j < 8; ++j) pk[j] = (__bf16)(acc[c][j] + bv_);
      *(bf16x8*)(Vt + ((size_t)bh * HDIM + dcol) * T_SEQ + trow) = pk;
    } else {                                        // Q/K: [BH][T][D]
      __bf16* dst = (which == 0 ? Qb : Kb);
      #pragma unroll
      for (int j = 0; j < 8; ++j)
        dst[((size_t)bh * T_SEQ + trow + j) * HDIM + dcol] =
            (__bf16)(acc[c][j] + bv_);
    }
  }
}

// ---------------------------------------------------------------------------
// Kernel 2: flash attention. grid = (T/256, B*NHEADS), block = 256 (8 waves).
// Each wave owns TWO 16-q tiles (u=0,1). K/V tiles (128 kv rows) staged into
// LDS by the TDM (wave 0 issues; double-buffered; s_wait_tensorcnt + barrier).
// All 8 waves * 2 q-tiles share each staged tile -> 16x less L2 traffic.
// ---------------------------------------------------------------------------
__global__ __launch_bounds__(256) void flash_attn_kernel(
    const __bf16* __restrict__ Qb, const __bf16* __restrict__ Kb,
    const __bf16* __restrict__ Vt, float* __restrict__ out)
{
  __shared__ __align__(16) __bf16 ldsK[2][KV_TILE * HDIM];   // [kv][d]  16KB x2
  __shared__ __align__(16) __bf16 ldsV[2][HDIM * KV_TILE];   // [d][kv]  16KB x2

  const int tid  = threadIdx.x;
  const int wave = tid >> 5;
  const int lane = tid & 31;
  const int l16  = lane & 15;
  const int hl   = lane >> 4;

  const int bh = blockIdx.y;
  const int b  = bh >> 3;
  const int h  = bh & 7;

  // Q fragments (B operand of S^T) for both q-tiles; reused all kv steps.
  v16bf qf[2][2];
  int qrow[2];
  #pragma unroll
  for (int u = 0; u < 2; ++u) {
    qrow[u] = blockIdx.x * 256 + (wave + 8 * u) * 16 + l16;
    const __bf16* qr = Qb + ((size_t)bh * T_SEQ + qrow[u]) * HDIM + 8 * hl;
    #pragma unroll
    for (int c = 0; c < 2; ++c) {
      bf16x8 lo = *(const bf16x8*)(qr + 32 * c);
      bf16x8 hi = *(const bf16x8*)(qr + 32 * c + 16);
      #pragma unroll
      for (int e = 0; e < 8; ++e) { qf[u][c][e] = lo[e]; qf[u][c][e+8] = hi[e]; }
    }
  }

  const __bf16* kbase = Kb + (size_t)bh * T_SEQ * HDIM;  // [T][64]
  const __bf16* vbase = Vt + (size_t)bh * HDIM * T_SEQ;  // [64][T]

  const v8f zero8 = {0.f,0.f,0.f,0.f,0.f,0.f,0.f,0.f};
  v8f o[2][4] = {{zero8,zero8,zero8,zero8},{zero8,zero8,zero8,zero8}};
  float run_max[2] = {-3.0e38f, -3.0e38f};
  float run_sum[2] = {0.f, 0.f};
  const float scale = 0.125f;                            // 1/sqrt(64)

  // Prime double buffer: TDM pulls K tile [128x64] and V^T tile [64x128].
  if (wave == 0) {
    tdm_load_tile_2d(lds_offset_of(&ldsK[0][0]), kbase,
                     HDIM, T_SEQ, HDIM, KV_TILE, HDIM);
    tdm_load_tile_2d(lds_offset_of(&ldsV[0][0]), vbase,
                     T_SEQ, HDIM, KV_TILE, HDIM, T_SEQ);
  }

  int buf = 0;
  for (int kv0 = 0; kv0 < T_SEQ; kv0 += KV_TILE, buf ^= 1) {
    if (wave == 0) __builtin_amdgcn_s_wait_tensorcnt(0); // tiles[buf] landed
    __syncthreads();                                     // visible to all waves
    if (wave == 0 && kv0 + KV_TILE < T_SEQ) {            // kick next transfer
      tdm_load_tile_2d(lds_offset_of(&ldsK[buf ^ 1][0]),
                       kbase + (size_t)(kv0 + KV_TILE) * HDIM,
                       HDIM, T_SEQ, HDIM, KV_TILE, HDIM);
      tdm_load_tile_2d(lds_offset_of(&ldsV[buf ^ 1][0]),
                       vbase + (kv0 + KV_TILE),
                       T_SEQ, HDIM, KV_TILE, HDIM, T_SEQ);
    }
    const __bf16* lk = &ldsK[buf][0];
    const __bf16* lv = &ldsV[buf][0];

    for (int kvl = 0; kvl < KV_TILE; kvl += 32) {
      // --- S^T: two 16(kv) x 16(q) fragments per q-tile, over d=64 ---
      float st[2][2][8];
      #pragma unroll
      for (int s = 0; s < 2; ++s) {
        const __bf16* kr = lk + (size_t)(kvl + 16 * s + l16) * HDIM + 8 * hl;
        v8f sacc0 = zero8, sacc1 = zero8;
        #pragma unroll
        for (int c = 0; c < 2; ++c) {
          bf16x8 lo = *(const bf16x8*)(kr + 32 * c);
          bf16x8 hi = *(const bf16x8*)(kr + 32 * c + 16);
          v16bf kf;
          #pragma unroll
          for (int e = 0; e < 8; ++e) { kf[e] = lo[e]; kf[e + 8] = hi[e]; }
          sacc0 = __builtin_amdgcn_wmma_f32_16x16x32_bf16(
              false, kf, false, qf[0][c], (short)0, sacc0, false, false);
          sacc1 = __builtin_amdgcn_wmma_f32_16x16x32_bf16(
              false, kf, false, qf[1][c], (short)0, sacc1, false, false);
        }
        #pragma unroll
        for (int j = 0; j < 8; ++j) {
          st[0][s][j] = sacc0[j] * scale;
          st[1][s][j] = sacc1[j] * scale;
        }
      }

      // --- online softmax per q-tile (per-lane scalars) ---
      v16bf pf[2];
      float fac[2];
      #pragma unroll
      for (int u = 0; u < 2; ++u) {
        float cm = st[u][0][0];
        #pragma unroll
        for (int j = 0; j < 8; ++j) {
          cm = fmaxf(cm, st[u][0][j]);
          cm = fmaxf(cm, st[u][1][j]);
        }
        cm = fmaxf(cm, __shfl_xor(cm, 16, 32));
        const float m_new = fmaxf(run_max[u], cm);
        fac[u] = __expf(run_max[u] - m_new);
        run_max[u] = m_new;
        float ls = 0.f;
        #pragma unroll
        for (int j = 0; j < 8; ++j) {
          float e0 = __expf(st[u][0][j] - m_new);
          float e1 = __expf(st[u][1][j] - m_new);
          ls += e0 + e1;
          pf[u][j]     = (__bf16)e0;
          pf[u][j + 8] = (__bf16)e1;
        }
        ls += __shfl_xor(ls, 16, 32);
        run_sum[u] = run_sum[u] * fac[u] + ls;
      }

      // --- O^T += V^T(32 kv slice) * P^T ; vf shared by both q-tiles ---
      #pragma unroll
      for (int f = 0; f < 4; ++f) {
        const __bf16* vr = lv + (size_t)(16 * f + l16) * KV_TILE + kvl + 8 * hl;
        bf16x8 lo = *(const bf16x8*)(vr);
        bf16x8 hi = *(const bf16x8*)(vr + 16);
        v16bf vf;
        #pragma unroll
        for (int e = 0; e < 8; ++e) { vf[e] = lo[e]; vf[e + 8] = hi[e]; }
        #pragma unroll
        for (int u = 0; u < 2; ++u) {
          #pragma unroll
          for (int j = 0; j < 8; ++j) o[u][f][j] *= fac[u];
          o[u][f] = __builtin_amdgcn_wmma_f32_16x16x32_bf16(
              false, vf, false, pf[u], (short)0, o[u][f], false, false);
        }
      }
    }
    __syncthreads();                  // all waves done with buf before reuse
  }

  // Normalize and store both q-tiles; d = 16f + 8*hl + j contiguous per lane.
  #pragma unroll
  for (int u = 0; u < 2; ++u) {
    const float inv = 1.0f / run_sum[u];
    float* orow = out + ((size_t)b * T_SEQ + qrow[u]) * MDIM + h * HDIM;
    #pragma unroll
    for (int f = 0; f < 4; ++f) {
      float4 w0 = make_float4(o[u][f][0]*inv, o[u][f][1]*inv,
                              o[u][f][2]*inv, o[u][f][3]*inv);
      float4 w1 = make_float4(o[u][f][4]*inv, o[u][f][5]*inv,
                              o[u][f][6]*inv, o[u][f][7]*inv);
      float* dst = orow + 16 * f + 8 * hl;
      *(float4*)(dst)     = w0;
      *(float4*)(dst + 4) = w1;
    }
  }
}

// ---------------------------------------------------------------------------
extern "C" void kernel_launch(void* const* d_in, const int* in_sizes, int n_in,
                              void* d_out, int out_size, void* d_ws, size_t ws_size,
                              hipStream_t stream) {
  (void)in_sizes; (void)n_in; (void)out_size; (void)ws_size;
  const float* x  = (const float*)d_in[0];
  const float* Wq = (const float*)d_in[1];
  const float* bq = (const float*)d_in[2];
  const float* Wk = (const float*)d_in[3];
  const float* bk = (const float*)d_in[4];
  const float* Wv = (const float*)d_in[5];
  const float* bv = (const float*)d_in[6];
  float* out = (float*)d_out;

  // Workspace (bf16): Q,K [BH,T,D] + V^T [BH,D,T] (8 MB each) + W^T (1.5 MB)
  const size_t elems = (size_t)BATCH * NHEADS * T_SEQ * HDIM;
  __bf16* Qb = (__bf16*)d_ws;
  __bf16* Kb = Qb + elems;
  __bf16* Vt = Kb + elems;
  __bf16* Wt = Vt + elems;

  prep_w_kernel<<<dim3(3 * W_ELEMS / 4 / 256), 256, 0, stream>>>(Wq, Wk, Wv, Wt);

  dim3 pg(BATCH * T_SEQ / 128, NHEADS, 3);
  qkv_proj_kernel<<<pg, 256, 0, stream>>>(x, bq, bk, bv, Wt, Qb, Kb, Vt);

  dim3 ag(T_SEQ / 256, BATCH * NHEADS);
  flash_attn_kernel<<<ag, 256, 0, stream>>>(Qb, Kb, Vt, out);
}
